// MultiHeadAttention_52690658787436
// MI455X (gfx1250) — matrix operand order repaired and checked by reading, not compiled, not run on hardware
//
#include <hip/hip_runtime.h>

// ---- problem constants (from reference) ----
constexpr int kB  = 4;
constexpr int kS  = 2048;
constexpr int kD  = 512;
constexpr int kH  = 8;
constexpr int kHD = 64;

typedef _Float16 half16 __attribute__((ext_vector_type(16)));
typedef float    float8 __attribute__((ext_vector_type(8)));

union FragH { half16 v; unsigned int u[8]; _Float16 h[16]; };

// A-matrix (16x32 f16) per-lane K index: VGPR v, lane-half sel
__device__ __forceinline__ int aKidx(int v, int half) {
    return (v >> 2) * 16 + half * 8 + (v & 3) * 2;
}
// B-matrix (32x16 f16) per-lane K index (same K order as A)
__device__ __forceinline__ int bKidx(int v, int half) {
    return (v >> 2) * 16 + half * 8 + (v & 3) * 2;
}

// per-lane async global->LDS copy, 16 bytes (ASYNCcnt-tracked, no VGPR round trip)
__device__ __forceinline__ void async_b128(unsigned lds_addr, const void* gp) {
    asm volatile("global_load_async_to_lds_b128 %0, %1, off"
                 :: "v"(lds_addr), "v"((unsigned long long)(size_t)gp)
                 : "memory");
}

// ---------------- f32 -> f16 convert (4 elems/thread) ----------------
__global__ void cvt_f32_f16(const float* __restrict__ in, _Float16* __restrict__ out, int n4) {
    int i = blockIdx.x * blockDim.x + threadIdx.x;
    if (i < n4) {
        float4 f = ((const float4*)in)[i];
        _Float16 h4[4] = {(_Float16)f.x, (_Float16)f.y, (_Float16)f.z, (_Float16)f.w};
        *(uint2*)(out + (size_t)i * 4) = *(const uint2*)h4;
    }
}

// ---------------- f32 [512][512] -> f16 transposed [512][512] ----------------
__global__ void cvt_transpose512(const float* __restrict__ in, _Float16* __restrict__ out) {
    int idx = blockIdx.x * 256 + threadIdx.x;   // 262144 elements
    int k = idx >> 9, n = idx & 511;
    out[(size_t)n * 512 + k] = (_Float16)in[idx];
}

// ---------------- GEMM: C = A[M,512] x W[512,512] + bias ----------------
// A f16 row-major; Wt = W^T f16 row-major (pre-transposed).
// 256 threads = 8 waves; block tile 128x128; wave tile 32x64.
// Double-buffered async (global->LDS) staging of both tiles.
// MODE 0: f16 head-split [B,H,S,64]   (Q, K)
// MODE 1: f16 head-split transposed [B,H,64,S]  (V^T)
// MODE 2: f32 row-major [M,512]       (final output)
template <int MODE>
__global__ __launch_bounds__(256) void gemm512(const _Float16* __restrict__ A,
                                               const _Float16* __restrict__ Wt,
                                               const float* __restrict__ bias,
                                               void* __restrict__ out,
                                               float scale) {
    __shared__ _Float16 As[2][128][40];   // 32 halves data, pitch 40 (16B-aligned, bank-safe)
    __shared__ _Float16 Bs[2][128][40];   // Bs[buf][n][k] = Wt rows = W columns

    const int tid  = threadIdx.x;
    const int lane = tid & 31;
    const int wv   = tid >> 5;
    const int half = lane >> 4;
    const int l16  = lane & 15;
    const int m0   = blockIdx.x * 128;
    const int n0   = blockIdx.y * 128;
    const int wrow = (wv & 3) * 32;    // wave M offset within block
    const int wcol = (wv >> 2) * 64;   // wave N offset within block
    const int r0   = tid >> 2;         // staging row 0..63 (+64 on rep 1)
    const int q0   = tid & 3;          // staging 16B segment

    // issue the 4 per-wave async b128 copies for one 32-wide k chunk
    auto issue = [&](int k0, int buf) {
        #pragma unroll
        for (int rep = 0; rep < 2; ++rep) {
            int row = r0 + rep * 64;
            async_b128((unsigned)(size_t)&As[buf][row][q0 * 8],
                       A  + (size_t)(m0 + row) * kD + k0 + q0 * 8);
            async_b128((unsigned)(size_t)&Bs[buf][row][q0 * 8],
                       Wt + (size_t)(n0 + row) * kD + k0 + q0 * 8);
        }
    };

    float8 acc[2][4];
    #pragma unroll
    for (int mi = 0; mi < 2; ++mi)
        #pragma unroll
        for (int t = 0; t < 4; ++t) acc[mi][t] = float8{};

    issue(0, 0);
    for (int i = 0; i < kD / 32; ++i) {
        const int cur = i & 1;
        __syncthreads();                       // everyone done reading buf cur^1
        if (i + 1 < kD / 32) {
            issue((i + 1) * 32, cur ^ 1);      // overlap DMA of next chunk with compute
            asm volatile("s_wait_asynccnt 4" ::: "memory");   // chunk i landed (in-order)
        } else {
            asm volatile("s_wait_asynccnt 0" ::: "memory");
        }
        __syncthreads();                       // all waves' chunk-i copies visible

        FragH af[2];
        #pragma unroll
        for (int mi = 0; mi < 2; ++mi)
            #pragma unroll
            for (int v = 0; v < 8; ++v)
                af[mi].u[v] = *(const unsigned int*)&As[cur][wrow + mi * 16 + l16][aKidx(v, half)];
        #pragma unroll
        for (int t = 0; t < 4; ++t) {
            FragH bf;
            #pragma unroll
            for (int v = 0; v < 8; ++v)
                bf.u[v] = *(const unsigned int*)&Bs[cur][wcol + t * 16 + l16][bKidx(v, half)];
            #pragma unroll
            for (int mi = 0; mi < 2; ++mi)
                acc[mi][t] = __builtin_amdgcn_wmma_f32_16x16x32_f16(false, af[mi].v, false, bf.v,
                                                                    (short)0, acc[mi][t], false, false);
        }
    }

    // epilogue: C layout -> lane holds col N=l16 (per half), VGPR v holds row M=v+8*half
    #pragma unroll
    for (int mi = 0; mi < 2; ++mi) {
        #pragma unroll
        for (int t = 0; t < 4; ++t) {
            int col = n0 + wcol + t * 16 + l16;
            float bsv = bias[col];
            #pragma unroll
            for (int v = 0; v < 8; ++v) {
                int gm  = m0 + wrow + mi * 16 + v + 8 * half;
                float val = (acc[mi][t][v] + bsv) * scale;
                if (MODE == 2) {
                    ((float*)out)[(size_t)gm * kD + col] = val;
                } else {
                    int b = gm >> 11, s = gm & (kS - 1);
                    int h = col >> 6, d = col & (kHD - 1);
                    _Float16* o16 = (_Float16*)out;
                    if (MODE == 0)
                        o16[(((size_t)(b * kH + h) * kS + s) * kHD) + d] = (_Float16)val;
                    else
                        o16[(((size_t)(b * kH + h) * kHD + d) * kS) + s] = (_Float16)val;
                }
            }
        }
    }
}

// ---------------- flash attention ----------------
// One wave per (b, h, 16-row q tile). 8 waves/block, fully independent.
// Qh, Kh: [B,H,S,64] f16 (Q pre-scaled by 1/8). Vt: [B,H,64,S] f16.
// Output Attn: [B,S,512] f16 (heads merged).
__global__ __launch_bounds__(256) void flash_attn(const _Float16* __restrict__ Qh,
                                                  const _Float16* __restrict__ Kh,
                                                  const _Float16* __restrict__ Vt,
                                                  _Float16* __restrict__ Attn) {
    __shared__ _Float16 Pl[8][16][40];   // per-wave P bounce, pitch 40

    const int lane  = threadIdx.x & 31;
    const int wv    = threadIdx.x >> 5;
    const int half  = lane >> 4;
    const int l16   = lane & 15;
    const int job   = blockIdx.x * 8 + wv;      // 512 blocks * 8 = 4096 = 32 bh * 128 qt
    const int bh    = job >> 7;
    const int qt    = job & 127;
    const int qbase = qt * 16;
    const int b     = bh >> 3, h = bh & 7;

    const _Float16* Qb = Qh + (size_t)bh * kS * kHD;
    const _Float16* Kb = Kh + (size_t)bh * kS * kHD;
    const _Float16* Vb = Vt + (size_t)bh * kHD * kS;

    // Q A-fragments (two 16x32 chunks of head dim), direct from global (contiguous pairs)
    FragH qf[2];
    #pragma unroll
    for (int c = 0; c < 2; ++c)
        #pragma unroll
        for (int v = 0; v < 8; ++v)
            qf[c].u[v] = *(const unsigned int*)(Qb + (size_t)(qbase + l16) * kHD + c * 32 + aKidx(v, half));

    float8 acc[4];
    #pragma unroll
    for (int t = 0; t < 4; ++t) acc[t] = float8{};
    float m[8], l[8];
    #pragma unroll
    for (int v = 0; v < 8; ++v) { m[v] = -1e30f; l[v] = 0.0f; }

    const int nsteps = (qbase + 16 + 31) >> 5;     // causal: key blocks with kbase <= qbase+15
    for (int k2 = 0; k2 < nsteps; ++k2) {
        const int kbase = k2 * 32;

        // prefetch next key block (K rows + V^T rows) -> global_prefetch_b8
        if (k2 + 1 < nsteps) {
            __builtin_prefetch(Kb + (size_t)(kbase + 32 + lane) * kHD, 0, 3);
            __builtin_prefetch(Vb + (size_t)(lane * 2) * kS + kbase + 32, 0, 3);
        }

        // scores: two 16x16 tiles (Q 16x64 x K^T 64x32), K rows are B-frags in-place
        float8 sc[2];
        #pragma unroll
        for (int nt = 0; nt < 2; ++nt) {
            const int keyrow = kbase + nt * 16 + l16;
            FragH kf0, kf1;
            #pragma unroll
            for (int v = 0; v < 8; ++v) {
                kf0.u[v] = *(const unsigned int*)(Kb + (size_t)keyrow * kHD +      bKidx(v, half));
                kf1.u[v] = *(const unsigned int*)(Kb + (size_t)keyrow * kHD + 32 + bKidx(v, half));
            }
            float8 c = float8{};
            c = __builtin_amdgcn_wmma_f32_16x16x32_f16(false, qf[0].v, false, kf0.v, (short)0, c, false, false);
            c = __builtin_amdgcn_wmma_f32_16x16x32_f16(false, qf[1].v, false, kf1.v, (short)0, c, false, false);
            sc[nt] = c;
        }

        // causal mask (only diagonal blocks need it)
        if (kbase + 31 > qbase) {
            #pragma unroll
            for (int nt = 0; nt < 2; ++nt) {
                int col = kbase + nt * 16 + l16;
                #pragma unroll
                for (int v = 0; v < 8; ++v) {
                    int row = qbase + v + 8 * half;
                    if (col > row) sc[nt][v] = -1e30f;
                }
            }
        }

        // online softmax per row (rows of C-frag: M = v + 8*half, cols across 16 lanes)
        #pragma unroll
        for (int v = 0; v < 8; ++v) {
            float s0 = sc[0][v], s1 = sc[1][v];
            float bm = fmaxf(s0, s1);
            #pragma unroll
            for (int off = 1; off < 16; off <<= 1)
                bm = fmaxf(bm, __shfl_xor(bm, off, 32));
            float nm   = fmaxf(m[v], bm);
            float corr = __expf(m[v] - nm);
            float p0   = __expf(s0 - nm);
            float p1   = __expf(s1 - nm);
            float rs   = p0 + p1;
            #pragma unroll
            for (int off = 1; off < 16; off <<= 1)
                rs += __shfl_xor(rs, off, 32);
            m[v] = nm;
            l[v] = l[v] * corr + rs;
            #pragma unroll
            for (int t = 0; t < 4; ++t) acc[t][v] *= corr;
            int row = v + 8 * half;
            Pl[wv][row][l16]      = (_Float16)p0;   // key tile 0
            Pl[wv][row][16 + l16] = (_Float16)p1;   // key tile 1
        }

        // per-wave LDS ordering: DS ops are in-order per wave; fence the compiler + DScnt
        asm volatile("s_wait_dscnt 0" ::: "memory");

        // P as A-fragment (16x32)
        FragH pf;
        #pragma unroll
        for (int v = 0; v < 8; ++v)
            pf.u[v] = *(const unsigned int*)&Pl[wv][l16][aKidx(v, half)];

        // PV: V^T rows are B-frags in-place (contiguous key pairs)
        #pragma unroll
        for (int t = 0; t < 4; ++t) {
            FragH vf;
            int d = t * 16 + l16;
            #pragma unroll
            for (int v = 0; v < 8; ++v)
                vf.u[v] = *(const unsigned int*)(Vb + (size_t)d * kS + kbase + bKidx(v, half));
            acc[t] = __builtin_amdgcn_wmma_f32_16x16x32_f16(false, pf.v, false, vf.v, (short)0, acc[t], false, false);
        }
        asm volatile("" ::: "memory");   // keep next-iter P writes after this iter's reads
    }

    // normalize and write merged-head f16: Attn[b, s, h*64 + d]
    #pragma unroll
    for (int t = 0; t < 4; ++t) {
        int col = h * kHD + t * 16 + l16;
        #pragma unroll
        for (int v = 0; v < 8; ++v) {
            int srow = qbase + v + 8 * half;
            float val = acc[t][v] / l[v];
            Attn[((size_t)(b * kS + srow)) * kD + col] = (_Float16)val;
        }
    }
}

// ---------------- launch ----------------
extern "C" void kernel_launch(void* const* d_in, const int* in_sizes, int n_in,
                              void* d_out, int out_size, void* d_ws, size_t ws_size,
                              hipStream_t stream) {
    (void)in_sizes; (void)n_in; (void)out_size; (void)ws_size;
    const float* X  = (const float*)d_in[0];
    const float* Wq = (const float*)d_in[1];
    const float* bq = (const float*)d_in[2];
    const float* Wk = (const float*)d_in[3];
    const float* bk = (const float*)d_in[4];
    const float* Wv = (const float*)d_in[5];
    const float* bv = (const float*)d_in[6];
    const float* Wo = (const float*)d_in[7];
    const float* bo = (const float*)d_in[8];

    const size_t NX = (size_t)kB * kS * kD;   // 4,194,304
    const size_t NW = (size_t)kD * kD;        //   262,144

    char* ws = (char*)d_ws;
    _Float16* Xh  = (_Float16*)ws; ws += NX * 2;
    _Float16* Wqt = (_Float16*)ws; ws += NW * 2;   // W^T, f16
    _Float16* Wkt = (_Float16*)ws; ws += NW * 2;
    _Float16* Wvt = (_Float16*)ws; ws += NW * 2;
    _Float16* Wot = (_Float16*)ws; ws += NW * 2;
    _Float16* Qh  = (_Float16*)ws; ws += NX * 2;
    _Float16* Kh  = (_Float16*)ws; ws += NX * 2;
    _Float16* Vt  = (_Float16*)ws; ws += NX * 2;
    _Float16* At  = (_Float16*)ws; ws += NX * 2;

    cvt_f32_f16<<<(unsigned)(NX / 4 / 256), 256, 0, stream>>>(X, Xh, (int)(NX / 4));
    cvt_transpose512<<<(unsigned)(NW / 256), 256, 0, stream>>>(Wq, Wqt);
    cvt_transpose512<<<(unsigned)(NW / 256), 256, 0, stream>>>(Wk, Wkt);
    cvt_transpose512<<<(unsigned)(NW / 256), 256, 0, stream>>>(Wv, Wvt);
    cvt_transpose512<<<(unsigned)(NW / 256), 256, 0, stream>>>(Wo, Wot);

    dim3 g((kB * kS) / 128, kD / 128), blk(256);
    gemm512<0><<<g, blk, 0, stream>>>(Xh, Wqt, bq, Qh, 0.125f);  // Q pre-scaled by 1/sqrt(64)
    gemm512<0><<<g, blk, 0, stream>>>(Xh, Wkt, bk, Kh, 1.0f);
    gemm512<1><<<g, blk, 0, stream>>>(Xh, Wvt, bv, Vt, 1.0f);    // V stored transposed

    flash_attn<<<512, 256, 0, stream>>>(Qh, Kh, Vt, At);

    gemm512<2><<<g, blk, 0, stream>>>(At, Wot, bo, d_out, 1.0f); // final projection, f32 out
}
